// TPAsyncDense_68255620268618
// MI455X (gfx1250) — compile-verified, matrix operand order checked by reading, not run on hardware
//
#include <hip/hip_runtime.h>

typedef __bf16 v16bf __attribute__((ext_vector_type(16)));
typedef __bf16 v8bf  __attribute__((ext_vector_type(8)));
typedef float  v8f   __attribute__((ext_vector_type(8)));
typedef int    v4i   __attribute__((ext_vector_type(4)));

typedef __attribute__((address_space(1))) v4i* gv4i_p;   // global int4*
typedef __attribute__((address_space(3))) v4i* lv4i_p;   // LDS int4*

#define DSH    4
#define MTOT   8192          // B*S
#define KTOT   2048          // D*DIN_S
#define NTOT   2048          // DOUT_S
#define KSHARD 512           // DIN_S
#define TILE_M 128           // block rows
#define TILE_N 256           // block cols
#define TILE_K 64            // staged K per pipeline stage (2 WMMA K-steps)
#define NKT    (KTOT / TILE_K)   // 32 k-tiles
#define LDK    72            // padded LDS row stride (144 B = 36 banks)

#if __has_builtin(__builtin_amdgcn_global_load_async_to_lds_b128)
#define ASYNC_LDS 1
#endif

__device__ __forceinline__ unsigned short f2bf(float f) {
    return __builtin_bit_cast(unsigned short, (__bf16)f);   // RNE convert
}
__device__ __forceinline__ unsigned pack2bf(float a, float b) {
    return (unsigned)f2bf(a) | ((unsigned)f2bf(b) << 16);
}

// 16-byte global -> LDS mover: async DMA on CDNA5, reg-staged otherwise.
__device__ __forceinline__ void g2l_b128(const unsigned short* g, unsigned short* l) {
#ifdef ASYNC_LDS
    __builtin_amdgcn_global_load_async_to_lds_b128((gv4i_p)g, (lv4i_p)l, 0, 0);
#else
    *(uint4*)l = *(const uint4*)g;
#endif
}
__device__ __forceinline__ void wait_lds_arrival() {
#ifdef ASYNC_LDS
#if __has_builtin(__builtin_amdgcn_s_wait_asynccnt)
    __builtin_amdgcn_s_wait_asynccnt(0);
#else
    asm volatile("s_wait_asynccnt 0x0" ::: "memory");
#endif
#endif
}

// ---------------------------------------------------------------------------
// Pre-pass 1: x fp32 -> bf16 copy (16,777,216 elements).
// ---------------------------------------------------------------------------
__global__ __launch_bounds__(256)
void convert_x_bf16(const float* __restrict__ x, unsigned short* __restrict__ xb) {
    const size_t base = ((size_t)blockIdx.x * 256 + threadIdx.x) * 8;
#pragma unroll
    for (int h = 0; h < 2; ++h) {
        const float4 v = *(const float4*)(x + base + h * 4);
        uint2 p;
        p.x = pack2bf(v.x, v.y);
        p.y = pack2bf(v.z, v.w);
        *(uint2*)(xb + base + h * 4) = p;
    }
}

// ---------------------------------------------------------------------------
// Pre-pass 2: W [d,i,k,n] fp32 -> Wt [d,i,n,k] bf16 (LDS-tiled 64x64 transpose).
// grid = (NTOT/64, KSHARD/64, DSH*DSH)
// ---------------------------------------------------------------------------
__global__ __launch_bounds__(256)
void transpose_w_bf16(const float* __restrict__ W, unsigned short* __restrict__ wt) {
    __shared__ unsigned short tile[64 * 72];

    const int t  = threadIdx.x;
    const int z  = blockIdx.z;
    const int n0 = blockIdx.x * 64;
    const int k0 = blockIdx.y * 64;

#pragma unroll
    for (int j = 0; j < 4; ++j) {
        const int f  = t + 256 * j;
        const int kr = f >> 4;
        const int nc = (f & 15) * 4;
        const float4 v = *(const float4*)(W +
            ((size_t)z * KSHARD + k0 + kr) * NTOT + n0 + nc);
        uint2 p;
        p.x = pack2bf(v.x, v.y);
        p.y = pack2bf(v.z, v.w);
        *(uint2*)&tile[kr * 72 + nc] = p;
    }
    __syncthreads();

#pragma unroll
    for (int j = 0; j < 4; ++j) {
        const int f  = t + 256 * j;
        const int nr = f >> 4;
        const int kc = (f & 15) * 4;
        uint2 p;
        p.x = (unsigned)tile[(kc + 0) * 72 + nr] | ((unsigned)tile[(kc + 1) * 72 + nr] << 16);
        p.y = (unsigned)tile[(kc + 2) * 72 + nr] | ((unsigned)tile[(kc + 3) * 72 + nr] << 16);
        *(uint2*)(wt + ((size_t)z * NTOT + n0 + nr) * KSHARD + k0 + kc) = p;
    }
}

// ---------------------------------------------------------------------------
// Main GEMM: 128x256 block tile, 8 waves of 64x64, double-buffered async LDS.
// ---------------------------------------------------------------------------
__global__ __launch_bounds__(256)
void tp_gemm_wmma_bf16(const unsigned short* __restrict__ xb, // [4,8192,512]
                       const unsigned short* __restrict__ wt, // [4,4,2048,512]
                       const float* __restrict__ bias,        // [4,2048]
                       float* __restrict__ out)               // [4,8192,2048]
{
    __shared__ unsigned short As[2 * TILE_M * LDK];   // 2 x 18 KB
    __shared__ unsigned short Bs[2 * TILE_N * LDK];   // 2 x 36 KB

    const int t    = threadIdx.x;
    const int wid  = t >> 5;
    const int lane = t & 31;
    const int half = lane >> 4;
    const int ln   = lane & 15;

    const int d  = blockIdx.z;
    const int m0 = blockIdx.y * TILE_M;
    const int n0 = blockIdx.x * TILE_N;

    const int waveM = wid & 1;    // 2 waves x 64 rows
    const int waveN = wid >> 1;   // 4 waves x 64 cols

    v8f acc[4][4];
#pragma unroll
    for (int am = 0; am < 4; ++am)
#pragma unroll
        for (int bn = 0; bn < 4; ++bn)
            acc[am][bn] = (v8f){0.f, 0.f, 0.f, 0.f, 0.f, 0.f, 0.f, 0.f};

    float bval[4];
#pragma unroll
    for (int bn = 0; bn < 4; ++bn)
        bval[bn] = bias[(size_t)d * NTOT + n0 + waveN * 64 + bn * 16 + ln];

    // staging decode: each thread moves 16B chunks; 8 threads cover a 64-k row
    const int srow = t >> 3;          // 0..31, step 32 per j
    const int sc8  = (t & 7) * 8;     // k offset in row

    const unsigned short* aBase = xb + (size_t)m0 * KSHARD;
    const unsigned short* bBase = wt + ((size_t)d * DSH * NTOT + n0) * KSHARD;

    // issue one k-tile's async loads into buffer `buf`
    auto stage = [&](int kt, int buf) {
        const int shard = kt >> 3;               // 8 k-tiles per shard
        const int kl    = (kt & 7) << 6;
        const unsigned short* aSrc = aBase + (size_t)shard * MTOT * KSHARD + kl + sc8;
        const unsigned short* bSrc = bBase + (size_t)shard * NTOT * KSHARD + kl + sc8;
        unsigned short* aDst = &As[buf * TILE_M * LDK + srow * LDK + sc8];
        unsigned short* bDst = &Bs[buf * TILE_N * LDK + srow * LDK + sc8];
#pragma unroll
        for (int j = 0; j < 4; ++j)
            g2l_b128(aSrc + (size_t)(srow + 32 * j) * KSHARD, aDst + 32 * j * LDK);
#pragma unroll
        for (int j = 0; j < 8; ++j)
            g2l_b128(bSrc + (size_t)(srow + 32 * j) * KSHARD, bDst + 32 * j * LDK);
    };

    stage(0, 0);

    for (int kt = 0; kt < NKT; ++kt) {
        const int cur = kt & 1;
        wait_lds_arrival();       // my async loads of tile kt landed in LDS
        __syncthreads();          // everyone's landed; everyone done with buf cur^1

        if (kt + 1 < NKT) stage(kt + 1, cur ^ 1);   // DMA next tile during compute

        const unsigned short* aT = &As[cur * TILE_M * LDK + (waveM * 64) * LDK];
        const unsigned short* bT = &Bs[cur * TILE_N * LDK + (waveN * 64) * LDK];

#pragma unroll
        for (int ks = 0; ks < TILE_K; ks += 32) {
            v16bf aF[4];
#pragma unroll
            for (int am = 0; am < 4; ++am) {
                const int row = am * 16 + ln;
                const v8bf lo = *(const v8bf*)&aT[row * LDK + ks + half * 8];
                const v8bf hi = *(const v8bf*)&aT[row * LDK + ks + 16 + half * 8];
                aF[am] = __builtin_shufflevector(lo, hi,
                    0,1,2,3,4,5,6,7,8,9,10,11,12,13,14,15);
            }
            v16bf bF[4];
#pragma unroll
            for (int bn = 0; bn < 4; ++bn) {
                const int row = bn * 16 + ln;
                const v8bf lo = *(const v8bf*)&bT[row * LDK + ks + half * 8];
                const v8bf hi = *(const v8bf*)&bT[row * LDK + ks + 16 + half * 8];
                bF[bn] = __builtin_shufflevector(lo, hi,
                    0,1,2,3,4,5,6,7,8,9,10,11,12,13,14,15);
            }
#pragma unroll
            for (int am = 0; am < 4; ++am)
#pragma unroll
                for (int bn = 0; bn < 4; ++bn)
                    acc[am][bn] = __builtin_amdgcn_wmma_f32_16x16x32_bf16(
                        false, aF[am], false, bF[bn],
                        (short)0, acc[am][bn], false, false);
        }
    }

#pragma unroll
    for (int am = 0; am < 4; ++am) {
#pragma unroll
        for (int bn = 0; bn < 4; ++bn) {
            const int n = n0 + waveN * 64 + bn * 16 + ln;
#pragma unroll
            for (int r = 0; r < 8; ++r) {
                const int m = m0 + waveM * 64 + am * 16 + half * 8 + r;
                __builtin_nontemporal_store(acc[am][bn][r] + bval[bn],
                    out + ((size_t)d * MTOT + m) * NTOT + n);
            }
        }
    }
}

// ---------------------------------------------------------------------------
// Fallback (ws too small): fused convert + GEMM, 128x128 tile, K=32.
// ---------------------------------------------------------------------------
__global__ __launch_bounds__(256)
void tp_gemm_fused(const float* __restrict__ x, const float* __restrict__ W,
                   const float* __restrict__ bias, float* __restrict__ out)
{
    __shared__ unsigned short As[128 * 32];
    __shared__ unsigned short Bs[128 * 32];

    const int t = threadIdx.x, wid = t >> 5, lane = t & 31;
    const int half = lane >> 4, ln = lane & 15;
    const int d = blockIdx.z, m0 = blockIdx.y * 128, n0 = blockIdx.x * 128;
    const int waveM = wid & 3, waveN = wid >> 2;

    v8f acc[2][4];
#pragma unroll
    for (int am = 0; am < 2; ++am)
#pragma unroll
        for (int bn = 0; bn < 4; ++bn)
            acc[am][bn] = (v8f){0.f, 0.f, 0.f, 0.f, 0.f, 0.f, 0.f, 0.f};

    float bval[4];
#pragma unroll
    for (int bn = 0; bn < 4; ++bn)
        bval[bn] = bias[(size_t)d * NTOT + n0 + waveN * 64 + bn * 16 + ln];

    for (int kb = 0; kb < KTOT; kb += 32) {
        const int shard = kb >> 9, kl = kb & (KSHARD - 1);
#pragma unroll
        for (int j = 0; j < 4; ++j) {
            const int f = t + 256 * j, row = f >> 3, col = (f & 7) * 4;
            const float4 v = *(const float4*)(x +
                ((size_t)shard * MTOT + (m0 + row)) * KSHARD + kl + col);
            uint2 p; p.x = pack2bf(v.x, v.y); p.y = pack2bf(v.z, v.w);
            *(uint2*)&As[row * 32 + col] = p;
        }
#pragma unroll
        for (int j = 0; j < 4; ++j) {
            const int f = t + 256 * j, krow = f >> 5, col = (f & 31) * 4;
            const float4 v = *(const float4*)(W +
                (((size_t)(d * DSH + shard)) * KSHARD + kl + krow) * NTOT + n0 + col);
            Bs[(col + 0) * 32 + krow] = f2bf(v.x);
            Bs[(col + 1) * 32 + krow] = f2bf(v.y);
            Bs[(col + 2) * 32 + krow] = f2bf(v.z);
            Bs[(col + 3) * 32 + krow] = f2bf(v.w);
        }
        __syncthreads();

        v16bf aF[2];
#pragma unroll
        for (int am = 0; am < 2; ++am) {
            const int row = waveM * 32 + am * 16 + ln;
            const v8bf lo = *(const v8bf*)&As[row * 32 + half * 8];
            const v8bf hi = *(const v8bf*)&As[row * 32 + 16 + half * 8];
            aF[am] = __builtin_shufflevector(lo, hi,
                0,1,2,3,4,5,6,7,8,9,10,11,12,13,14,15);
        }
        v16bf bF[4];
#pragma unroll
        for (int bn = 0; bn < 4; ++bn) {
            const int row = waveN * 64 + bn * 16 + ln;
            const v8bf lo = *(const v8bf*)&Bs[row * 32 + half * 8];
            const v8bf hi = *(const v8bf*)&Bs[row * 32 + 16 + half * 8];
            bF[bn] = __builtin_shufflevector(lo, hi,
                0,1,2,3,4,5,6,7,8,9,10,11,12,13,14,15);
        }
#pragma unroll
        for (int am = 0; am < 2; ++am)
#pragma unroll
            for (int bn = 0; bn < 4; ++bn)
                acc[am][bn] = __builtin_amdgcn_wmma_f32_16x16x32_bf16(
                    false, aF[am], false, bF[bn],
                    (short)0, acc[am][bn], false, false);
        __syncthreads();
    }

#pragma unroll
    for (int am = 0; am < 2; ++am)
#pragma unroll
        for (int bn = 0; bn < 4; ++bn) {
            const int n = n0 + waveN * 64 + bn * 16 + ln;
#pragma unroll
            for (int r = 0; r < 8; ++r) {
                const int m = m0 + waveM * 32 + am * 16 + half * 8 + r;
                __builtin_nontemporal_store(acc[am][bn][r] + bval[bn],
                    out + ((size_t)d * MTOT + m) * NTOT + n);
            }
        }
}

extern "C" void kernel_launch(void* const* d_in, const int* in_sizes, int n_in,
                              void* d_out, int out_size, void* d_ws, size_t ws_size,
                              hipStream_t stream) {
    (void)in_sizes; (void)n_in; (void)out_size;
    const float* x  = (const float*)d_in[0];
    const float* W  = (const float*)d_in[1];
    const float* b  = (const float*)d_in[2];
    float* out = (float*)d_out;

    const size_t xb_elems = (size_t)DSH * MTOT * KSHARD;
    const size_t wt_elems = (size_t)DSH * DSH * NTOT * KSHARD;
    const size_t need = (xb_elems + wt_elems) * sizeof(unsigned short); // 64 MB

    if (ws_size >= need) {
        unsigned short* xbuf = (unsigned short*)d_ws;
        unsigned short* wbuf = xbuf + xb_elems;
        convert_x_bf16<<<xb_elems / (256 * 8), 256, 0, stream>>>(x, xbuf);
        transpose_w_bf16<<<dim3(NTOT / 64, KSHARD / 64, DSH * DSH), 256, 0, stream>>>(W, wbuf);
        tp_gemm_wmma_bf16<<<dim3(NTOT / TILE_N, MTOT / TILE_M, DSH), 256, 0, stream>>>(
            xbuf, wbuf, b, out);
    } else {
        tp_gemm_fused<<<dim3(NTOT / 128, MTOT / 128, DSH), 256, 0, stream>>>(x, W, b, out);
    }
}
